// GMM_Batch_40132174413878
// MI455X (gfx1250) — matrix operand order repaired and checked by reading, not compiled, not run on hardware
//
#include <hip/hip_runtime.h>

#define NB   16      // batch
#define NC   256     // channels
#define NPIX 16384   // H*W
#define NK   64      // prototypes

typedef __attribute__((ext_vector_type(16))) _Float16 v16h;
typedef __attribute__((ext_vector_type(8)))  _Float16 v8h;
typedef __attribute__((ext_vector_type(8)))  float    v8f;
typedef __attribute__((ext_vector_type(4)))  float    v4f;

__device__ __forceinline__ v8f wmma_f16(v16h a, v16h b, v8f c) {
    // D = A(16x32 f16) * B(32x16 f16) + C(16x16 f32)
    return __builtin_amdgcn_wmma_f32_16x16x32_f16(
        /*neg_a=*/false, a, /*neg_b=*/false, b,
        /*c_mod=*/(short)0, c, /*reuse_a=*/false, /*reuse_b=*/false);
}

// ---------------------------------------------------------------------------
// Assignment: Z[b][n][k] = softmax_k( scale * sum_c x[b][c][n] * M[b][k][c] )
//   FINAL=false: scale=20, store Z^T as f16 [b][k][n] (operand for update GEMM)
//   FINAL=true : scale=1,  store f32 softmax to out [b][n][k]
//   WRITEX16  : while streaming f32 x, also emit f16 mirror x16[b][c][n]
//   READX16   : source the staging from the f16 mirror (half the HBM bytes)
// Block = 128 threads (4 waves); block handles one batch x 64 pixels.
// ---------------------------------------------------------------------------
template <bool FINAL, bool WRITEX16, bool READX16>
__global__ __launch_bounds__(128) void em_assign(
    const float* __restrict__ x, _Float16* x16,
    const _Float16* __restrict__ mproto, _Float16* zt, float* out)
{
    __shared__ _Float16 Mlds[NK][NC + 8];   // prototypes [k][c], padded (33 KB)
    __shared__ _Float16 Xlds[64][40];       // x chunk transposed [n][c32], padded
    __shared__ _Float16 Zlds[4][NK][16];    // per-wave Z^T staging (8 KB)

    const int b    = blockIdx.y;
    const int n0   = blockIdx.x * 64;
    const int tid  = threadIdx.x;
    const int wave = tid >> 5;
    const int lane = tid & 31;
    const int hf   = lane >> 4;   // half of wave
    const int l15  = lane & 15;

    // stage prototypes for this batch: straight copy with padding
    {
        const v8h* src = (const v8h*)(mproto + (size_t)b * NK * NC);
        for (int i = tid; i < NK * NC / 8; i += 128) {
            int k  = i >> 5;          // 32 8-half chunks per 256-wide row
            int c8 = i & 31;
            *(v8h*)&Mlds[k][c8 * 8] = src[i];
        }
    }

    const v8f vzero = {0.f, 0.f, 0.f, 0.f, 0.f, 0.f, 0.f, 0.f};
    v8f acc[4] = {vzero, vzero, vzero, vzero};

    for (int cs = 0; cs < NC; cs += 32) {
        __syncthreads();
        // stage x[b][cs..cs+31][n0..n0+63] -> Xlds[n][c] (transpose, f16)
        {
            const int cl   = tid & 31;
            const int noff = (tid >> 5) << 4;
            const size_t rowoff = ((size_t)b * NC + cs + cl) * NPIX + n0 + noff;
            if constexpr (READX16) {
                const v8h* xp = (const v8h*)(x16 + rowoff);
                v8h h0 = xp[0], h1 = xp[1];
                #pragma unroll
                for (int e = 0; e < 8; ++e) {
                    Xlds[noff + e][cl]     = h0[e];
                    Xlds[noff + 8 + e][cl] = h1[e];
                }
            } else {
                const v4f* xp4 = (const v4f*)(x + rowoff);
                v8h h0, h1;
                #pragma unroll
                for (int q = 0; q < 4; ++q) {
                    v4f f = xp4[q];
                    #pragma unroll
                    for (int e = 0; e < 4; ++e) {
                        _Float16 hv = (_Float16)f[e];
                        Xlds[noff + q * 4 + e][cl] = hv;
                        if (q < 2) h0[q * 4 + e] = hv;
                        else       h1[(q - 2) * 4 + e] = hv;
                    }
                }
                if constexpr (WRITEX16) {
                    v8h* dst = (v8h*)(x16 + rowoff);
                    dst[0] = h0; dst[1] = h1;
                }
            }
        }
        __syncthreads();

        // A fragment: 16 pixel-rows of this wave, K-dim = 32 channels.
        // k order used here (same for B): elems 0..15 -> K = 16*hf + 0..15 (contiguous).
        v16h a;
        {
            v8h lo = *(const v8h*)&Xlds[wave * 16 + l15][hf * 16];
            v8h hi = *(const v8h*)&Xlds[wave * 16 + l15][hf * 16 + 8];
            #pragma unroll
            for (int e = 0; e < 8; ++e) { a[e] = lo[e]; a[e + 8] = hi[e]; }
        }
        // B fragments: 4 tiles of 16 prototypes. k order as for A:
        // elems 0..15 -> K = 16*hf + 0..15 (contiguous), col = l15.
        #pragma unroll
        for (int t = 0; t < 4; ++t) {
            v8h lo = *(const v8h*)&Mlds[t * 16 + l15][cs + hf * 16];
            v8h hi = *(const v8h*)&Mlds[t * 16 + l15][cs + hf * 16 + 8];
            v16h bb;
            #pragma unroll
            for (int e = 0; e < 8; ++e) { bb[e] = lo[e]; bb[e + 8] = hi[e]; }
            acc[t] = wmma_f16(a, bb, acc[t]);
        }
    }

    // D layout: acc[t][j] = Z[row = j + 8*hf][col = 16*t + l15].
    // Row softmax over 64 cols: in-lane max/sum over t, then xor-butterfly
    // over masks 1,2,4,8 (stays within each 16-lane half = one row set).
    const float scale = FINAL ? 1.0f : 20.0f;
    #pragma unroll
    for (int j = 0; j < 8; ++j) {
        float mx = acc[0][j];
        #pragma unroll
        for (int t = 1; t < 4; ++t) mx = fmaxf(mx, acc[t][j]);
        #pragma unroll
        for (int m = 1; m <= 8; m <<= 1) mx = fmaxf(mx, __shfl_xor(mx, m, 32));
        float s = 0.f;
        #pragma unroll
        for (int t = 0; t < 4; ++t) {
            float e = __expf(scale * (acc[t][j] - mx));
            acc[t][j] = e; s += e;
        }
        #pragma unroll
        for (int m = 1; m <= 8; m <<= 1) s += __shfl_xor(s, m, 32);
        float inv = 1.0f / s;
        #pragma unroll
        for (int t = 0; t < 4; ++t) acc[t][j] *= inv;
    }

    if constexpr (FINAL) {
        #pragma unroll
        for (int j = 0; j < 8; ++j) {
            size_t row = (size_t)b * NPIX + n0 + wave * 16 + j + 8 * hf;
            #pragma unroll
            for (int t = 0; t < 4; ++t)
                out[row * NK + t * 16 + l15] = acc[t][j];
        }
    } else {
        // stage Z^T [k][n16] in LDS, then coalesced f16 stores [b][k][n]
        #pragma unroll
        for (int t = 0; t < 4; ++t)
            #pragma unroll
            for (int j = 0; j < 8; ++j)
                Zlds[wave][t * 16 + l15][j + 8 * hf] = (_Float16)acc[t][j];
        __syncthreads();
        #pragma unroll
        for (int r = 0; r < 2; ++r) {
            int k = r * 32 + lane;
            _Float16* dst = zt + ((size_t)b * NK + k) * NPIX + n0 + wave * 16;
            ((v8h*)dst)[0] = *(const v8h*)&Zlds[wave][k][0];
            ((v8h*)dst)[1] = *(const v8h*)&Zlds[wave][k][8];
        }
    }
}

// ---------------------------------------------------------------------------
// colsum[b][k] = sum_n Z[b][k][n]   (l1norm denominator, deterministic)
// ---------------------------------------------------------------------------
__global__ __launch_bounds__(256) void em_colsum(
    const _Float16* __restrict__ zt, float* __restrict__ colsum)
{
    __shared__ float red[256];
    const int b = blockIdx.y, k = blockIdx.x, t = threadIdx.x;
    const v8h* zr = (const v8h*)(zt + ((size_t)b * NK + k) * NPIX + t * 64);
    float s = 0.f;
    #pragma unroll
    for (int q = 0; q < 8; ++q) {
        v8h v = zr[q];
        #pragma unroll
        for (int e = 0; e < 8; ++e) s += (float)v[e];
    }
    red[t] = s;
    __syncthreads();
    for (int w = 128; w > 0; w >>= 1) {
        if (t < w) red[t] += red[t + w];
        __syncthreads();
    }
    if (t == 0) colsum[b * NK + k] = red[0];
}

// ---------------------------------------------------------------------------
// Update GEMM: Mpart[p][b][k][c] = sum_{n in chunk p} Z[b][k][n] * x[b][c][n]
// Block = (n-chunk p, batch) : 4 waves, wave owns 16 k-rows x all 256 c.
// ---------------------------------------------------------------------------
template <bool READX16>
__global__ __launch_bounds__(128) void em_update(
    const float* __restrict__ x, const _Float16* __restrict__ x16,
    const _Float16* __restrict__ zt, float* __restrict__ mpart)
{
    const int b    = blockIdx.y;
    const int n1   = blockIdx.x * (NPIX / 16);
    const int wave = threadIdx.x >> 5;
    const int lane = threadIdx.x & 31;
    const int hf   = lane >> 4;
    const int l15  = lane & 15;

    const v8f vzero = {0.f, 0.f, 0.f, 0.f, 0.f, 0.f, 0.f, 0.f};
    v8f acc[16];
    #pragma unroll
    for (int t = 0; t < 16; ++t) acc[t] = vzero;

    const _Float16* zrow = zt + ((size_t)b * NK + wave * 16 + l15) * NPIX;

    for (int n = n1; n < n1 + NPIX / 16; n += 32) {
        // A: Z rows (k), K-dim = 32 pixels, n-contiguous f16
        v16h a;
        {
            v8h lo = *(const v8h*)(zrow + n + hf * 16);
            v8h hi = *(const v8h*)(zrow + n + hf * 16 + 8);
            #pragma unroll
            for (int e = 0; e < 8; ++e) { a[e] = lo[e]; a[e + 8] = hi[e]; }
        }
        // B: x columns (c), K-dim = pixels (waves in block share via cache)
        #pragma unroll
        for (int t = 0; t < 16; ++t) {
            const size_t rowoff =
                ((size_t)b * NC + t * 16 + l15) * NPIX + n + hf * 16;
            v16h bb;
            if constexpr (READX16) {
                const v8h* xr = (const v8h*)(x16 + rowoff);
                v8h lo = xr[0], hi = xr[1];
                #pragma unroll
                for (int e = 0; e < 8; ++e) { bb[e] = lo[e]; bb[e + 8] = hi[e]; }
            } else {
                const v4f* xr4 = (const v4f*)(x + rowoff);
                #pragma unroll
                for (int q = 0; q < 4; ++q) {
                    v4f f = xr4[q];
                    #pragma unroll
                    for (int e = 0; e < 4; ++e) bb[q * 4 + e] = (_Float16)f[e];
                }
            }
            acc[t] = wmma_f16(a, bb, acc[t]);
        }
    }

    float* dst = mpart + (((size_t)blockIdx.x * NB + b) * NK) * NC;
    #pragma unroll
    for (int t = 0; t < 16; ++t)
        #pragma unroll
        for (int j = 0; j < 8; ++j)
            dst[(size_t)(wave * 16 + j + 8 * hf) * NC + t * 16 + l15] = acc[t][j];
}

// ---------------------------------------------------------------------------
// Normalize: M[b][k][c] = l2norm_c( (sum_p Mpart) / (eps + colsum[b][k]) )
// ---------------------------------------------------------------------------
__global__ __launch_bounds__(256) void em_normalize(
    const float* __restrict__ mpart, const float* __restrict__ colsum,
    _Float16* __restrict__ mproto)
{
    __shared__ float red[256];
    const int b = blockIdx.y, k = blockIdx.x, c = threadIdx.x;
    const size_t base = ((size_t)b * NK + k) * NC;

    float v = 0.f;
    #pragma unroll
    for (int p = 0; p < 16; ++p)
        v += mpart[(((size_t)p * NB + b) * NK + k) * NC + c];
    v /= (1e-6f + colsum[b * NK + k]);

    red[c] = v * v;
    __syncthreads();
    for (int w = 128; w > 0; w >>= 1) {
        if (c < w) red[c] += red[c + w];
        __syncthreads();
    }
    const float nrm = sqrtf(red[0]);
    mproto[base + c] = (_Float16)(v / (1e-6f + nrm));
}

// ---------------------------------------------------------------------------
// Init: broadcast mu [C,K] (already l2-normalized) to f16 M[b][k][c]
// ---------------------------------------------------------------------------
__global__ __launch_bounds__(256) void em_init(
    const float* __restrict__ mu, _Float16* __restrict__ mproto)
{
    const int b = blockIdx.y, k = blockIdx.x, c = threadIdx.x;
    mproto[((size_t)b * NK + k) * NC + c] = (_Float16)mu[c * NK + k];
}

extern "C" void kernel_launch(void* const* d_in, const int* in_sizes, int n_in,
                              void* d_out, int out_size, void* d_ws, size_t ws_size,
                              hipStream_t stream)
{
    (void)in_sizes; (void)n_in; (void)out_size;
    const float* x  = (const float*)d_in[0];   // [B, C, H*W] f32
    const float* mu = (const float*)d_in[1];   // [C, K] f32
    float*     out = (float*)d_out;            // [B, N, K] f32 (64 MiB)
    _Float16*  zt  = (_Float16*)d_out;         // Z^T f16 scratch (32 MiB of out)

    char* ws = (char*)d_ws;
    float*    colsum = (float*)ws;                                    //   4 KiB
    float*    mpart  = (float*)(ws + 4096);                           //  16 MiB
    _Float16* mproto = (_Float16*)(ws + 4096 + (size_t)16 * NB * NK * NC * 4);

    const size_t x16_off   = 4096 + (size_t)16 * NB * NK * NC * 4
                                  + (size_t)NB * NK * NC * 2;         // ~16.5 MiB
    const size_t x16_bytes = (size_t)NB * NC * NPIX * 2;              // 128 MiB
    _Float16* x16 = (_Float16*)(ws + x16_off);
    const bool fast = (ws_size >= x16_off + x16_bytes);               // constant

    dim3 gKC(NK, NB);            // (k, b)
    dim3 gA(NPIX / 64, NB);      // assignment tiles
    dim3 gU(16, NB);             // update n-chunks

    em_init<<<gKC, NC, 0, stream>>>(mu, mproto);
    if (fast) {
        // iter 0: stream f32 x once, emit f16 mirror as a side effect
        em_assign<false, true, false><<<gA, 128, 0, stream>>>(x, x16, mproto, zt, out);
        em_colsum<<<gKC, NC, 0, stream>>>(zt, colsum);
        em_update<true><<<gU, 128, 0, stream>>>(x, x16, zt, mpart);
        em_normalize<<<gKC, NC, 0, stream>>>(mpart, colsum, mproto);
        for (int it = 1; it < 3; ++it) {
            em_assign<false, false, true><<<gA, 128, 0, stream>>>(x, x16, mproto, zt, out);
            em_colsum<<<gKC, NC, 0, stream>>>(zt, colsum);
            em_update<true><<<gU, 128, 0, stream>>>(x, x16, zt, mpart);
            em_normalize<<<gKC, NC, 0, stream>>>(mpart, colsum, mproto);
        }
        em_assign<true, false, true><<<gA, 128, 0, stream>>>(x, x16, mproto, zt, out);
    } else {
        // fallback: small workspace, stream f32 x every pass
        for (int it = 0; it < 3; ++it) {
            em_assign<false, false, false><<<gA, 128, 0, stream>>>(x, x16, mproto, zt, out);
            em_colsum<<<gKC, NC, 0, stream>>>(zt, colsum);
            em_update<false><<<gU, 128, 0, stream>>>(x, x16, zt, mpart);
            em_normalize<<<gKC, NC, 0, stream>>>(mpart, colsum, mproto);
        }
        em_assign<true, false, false><<<gA, 128, 0, stream>>>(x, x16, mproto, zt, out);
    }
}
